// HyperbolicTopKTripletLoss_75136157876228
// MI455X (gfx1250) — matrix operand order repaired and checked by reading, not compiled, not run on hardware
//
#include <hip/hip_runtime.h>
#include <math.h>

#define B_N   8192
#define D_DIM 64
#define TOPK  5
#define NEGS  20
#define TAU_F 0.1f
#define EPSF  1e-9f
#define NTILE (B_N / 16)

typedef __attribute__((ext_vector_type(16))) _Float16 v16h;
typedef __attribute__((ext_vector_type(8)))  float    v8f;
typedef __attribute__((ext_vector_type(4)))  float    v4f;

// wave-relative LDS byte offset of a __shared__ object (addrspacecast -> ptrtoint)
__device__ __forceinline__ unsigned lds_off(const void* p) {
    return (unsigned)(uintptr_t)(__attribute__((address_space(3))) void*)(void*)p;
}

// ---------------------------------------------------------------------------
// Kernel 1 (pre-pass): per-row squared norm, 1/(1-||x||^2) factor, and the
// error-compensated f16 hi/lo split of x (computed ONCE, reused by all waves).
// ---------------------------------------------------------------------------
__global__ void prep_kernel(const float* __restrict__ x,
                            _Float16* __restrict__ xh, _Float16* __restrict__ xl,
                            float* __restrict__ nsq, float* __restrict__ inv) {
    int i = blockIdx.x * blockDim.x + threadIdx.x;
    if (i >= B_N) return;
    const float* p = x + (size_t)i * D_DIM;
    float s = 0.f;
#pragma unroll
    for (int c = 0; c < 4; ++c) {
        v16h vh, vl;
#pragma unroll
        for (int e = 0; e < 16; ++e) {
            float f = p[c * 16 + e];
            s = fmaf(f, f, s);
            _Float16 h = (_Float16)f;
            vh[e] = h;
            vl[e] = (_Float16)(f - (float)h);
        }
        *(v16h*)(xh + (size_t)i * D_DIM + c * 16) = vh;
        *(v16h*)(xl + (size_t)i * D_DIM + c * 16) = vl;
    }
    float cl = fminf(s, 1.f - EPSF);
    nsq[i] = s;
    inv[i] = 1.f / fmaxf(1.f - cl, EPSF);
}

// ---------------------------------------------------------------------------
// threefry2x32 (20 rounds), matching jax.random.key(42) -> key = (0, 42)
// ---------------------------------------------------------------------------
__device__ __forceinline__ unsigned rotl32(unsigned x, int r) {
    return (x << r) | (x >> (32 - r));
}

__device__ __forceinline__ void threefry2x32(unsigned k0, unsigned k1,
                                             unsigned x0, unsigned x1,
                                             unsigned& o0, unsigned& o1) {
    const unsigned ks2 = k0 ^ k1 ^ 0x1BD11BDAu;
    unsigned y0 = x0 + k0, y1 = x1 + k1;
    const int RA[4] = {13, 15, 26, 6};
    const int RB[4] = {17, 29, 16, 24};
#pragma unroll
    for (int i = 0; i < 4; ++i) { y0 += y1; y1 = rotl32(y1, RA[i]); y1 ^= y0; }
    y0 += k1;  y1 += ks2 + 1u;
#pragma unroll
    for (int i = 0; i < 4; ++i) { y0 += y1; y1 = rotl32(y1, RB[i]); y1 ^= y0; }
    y0 += ks2; y1 += k0 + 2u;
#pragma unroll
    for (int i = 0; i < 4; ++i) { y0 += y1; y1 = rotl32(y1, RA[i]); y1 ^= y0; }
    y0 += k0;  y1 += k1 + 3u;
#pragma unroll
    for (int i = 0; i < 4; ++i) { y0 += y1; y1 = rotl32(y1, RB[i]); y1 ^= y0; }
    y0 += k1;  y1 += ks2 + 4u;
#pragma unroll
    for (int i = 0; i < 4; ++i) { y0 += y1; y1 = rotl32(y1, RA[i]); y1 ^= y0; }
    y0 += ks2; y1 += k0 + 5u;
    o0 = y0; o1 = y1;
}

// ---------------------------------------------------------------------------
// Kernel 2: fused WMMA Gram + Poincare z + streaming top-5 + triplet loss.
// Block = 128 threads = 4 waves. Each wave owns a 16-row strip; the block
// double-buffers shared B tiles (hi+lo f16, 4 KB/tile) into LDS with
// global_load_async_to_lds_b128 (ASYNCcnt), so each tile is fetched once per
// block instead of once per wave. 6 x v_wmma_f32_16x16x32_f16 per tile in
// 3 independent 2-deep accumulator chains (kills WMMA->WMMA hazard NOPs).
// ---------------------------------------------------------------------------
__global__ void __launch_bounds__(128)
fused_kernel(const float* __restrict__ x,
             const _Float16* __restrict__ xh, const _Float16* __restrict__ xl,
             const float* __restrict__ nsq, const float* __restrict__ inv,
             float* __restrict__ loss) {
    // [buf][part(0=hi,1=lo)][col][elem]
    __shared__ __align__(64) _Float16 s_tile[2][2][16][D_DIM];
    __shared__ float    s_z[64][16][TOPK];
    __shared__ unsigned s_i[64][16][TOPK];

    const int tid   = threadIdx.x;
    const int lane  = tid & 31;
    const int wave  = tid >> 5;
    const int hiH   = lane >> 4;      // lane-half (K interleave select)
    const int l15   = lane & 15;
    const int rbase = blockIdx.x * 64 + wave * 16;

    // async-issue mapping: 128 threads x 32B cover one 4 KB tile (hi+lo)
    const int a_part = tid >> 6;          // 0 -> xh, 1 -> xl
    const int a_col  = (tid & 63) >> 2;   // 0..15
    const int a_q    = tid & 3;           // 32B quarter of a 128B row

    // ---- A tiles (16 rows x K=64), K order used here (same for B):
    // contiguous K: lanes 0-15: e=0..15 -> K=0..15 (lane-half adds +16)
    const _Float16* aph = xh + (size_t)(rbase + l15) * D_DIM + hiH * 16;
    const _Float16* apl = xl + (size_t)(rbase + l15) * D_DIM + hiH * 16;
    v16h a0h, a0l, a1h, a1l;
#pragma unroll
    for (int e = 0; e < 16; ++e) {
        int k0 = e;
        a0h[e] = aph[k0];      a0l[e] = apl[k0];
        a1h[e] = aph[k0 + 32]; a1l[e] = apl[k0 + 32];
    }

    // Per-lane row factors for the 8 C-matrix rows this lane holds
    float nr[8], ir[8];
#pragma unroll
    for (int v = 0; v < 8; ++v) {
        nr[v] = nsq[rbase + 8 * hiH + v];
        ir[v] = inv[rbase + 8 * hiH + v];
    }

    // Streaming per-lane top-5 (sorted ascending) per C-row
    float    tz[8][TOPK];
    unsigned ti[8][TOPK];
#pragma unroll
    for (int v = 0; v < 8; ++v)
#pragma unroll
        for (int j = 0; j < TOPK; ++j) { tz[v][j] = INFINITY; ti[v][j] = 0u; }

    const int diag_ct = rbase >> 4;   // the single tile containing diagonals

    // ---- prologue: async-stage tile 0 into buffer 0
    {
        const _Float16* src = (a_part ? xl : xh) + (size_t)a_col * D_DIM + a_q * 16;
        unsigned loff = lds_off(&s_tile[0][a_part][a_col][a_q * 16]);
        unsigned long long ga = (unsigned long long)(uintptr_t)src;
        asm volatile("global_load_async_to_lds_b128 %0, %1, off\n\t"
                     "global_load_async_to_lds_b128 %0, %1, off offset:16"
                     :: "v"(loff), "v"(ga) : "memory");
    }

    for (int ct = 0; ct < NTILE; ++ct) {
        const int cur = ct & 1;
        const int col = ct * 16 + l15;

        // own async loads for tile ct complete ...
        asm volatile("s_wait_asynccnt 0x0" ::: "memory");
        // ... then all waves' writes visible / previous readers drained
        __syncthreads();

        // stage tile ct+1 into the other buffer (safe only after the barrier)
        if (ct + 1 < NTILE) {
            const _Float16* src = (a_part ? xl : xh)
                                + (size_t)((ct + 1) * 16 + a_col) * D_DIM + a_q * 16;
            unsigned loff = lds_off(&s_tile[cur ^ 1][a_part][a_col][a_q * 16]);
            unsigned long long ga = (unsigned long long)(uintptr_t)src;
            asm volatile("global_load_async_to_lds_b128 %0, %1, off\n\t"
                         "global_load_async_to_lds_b128 %0, %1, off offset:16"
                         :: "v"(loff), "v"(ga) : "memory");
        }

        // ---- consume tile ct from LDS (ds_load_b128 pairs)
        v16h b0h = *(const v16h*)&s_tile[cur][0][l15][hiH * 16];
        v16h b1h = *(const v16h*)&s_tile[cur][0][l15][32 + hiH * 16];
        v16h b0l = *(const v16h*)&s_tile[cur][1][l15][hiH * 16];
        v16h b1l = *(const v16h*)&s_tile[cur][1][l15][32 + hiH * 16];

        // 3 independent 2-deep chains: hh, hi*lo, lo*hi
        v8f acc0 = {}, acc1 = {}, acc2 = {};
        acc0 = __builtin_amdgcn_wmma_f32_16x16x32_f16(false, a0h, false, b0h, (short)0, acc0, false, false);
        acc1 = __builtin_amdgcn_wmma_f32_16x16x32_f16(false, a0h, false, b0l, (short)0, acc1, false, false);
        acc2 = __builtin_amdgcn_wmma_f32_16x16x32_f16(false, a0l, false, b0h, (short)0, acc2, false, false);
        acc0 = __builtin_amdgcn_wmma_f32_16x16x32_f16(false, a1h, false, b1h, (short)0, acc0, false, false);
        acc1 = __builtin_amdgcn_wmma_f32_16x16x32_f16(false, a1h, false, b1l, (short)0, acc1, false, false);
        acc2 = __builtin_amdgcn_wmma_f32_16x16x32_f16(false, a1l, false, b1h, (short)0, acc2, false, false);

        const float ncj = nsq[col];
        const float ic  = inv[col];

        float zv[8];
#pragma unroll
        for (int v = 0; v < 8; ++v) {
            float g   = (acc0[v] + acc1[v]) + acc2[v];
            float dsq = fmaxf(fmaf(-2.f, g, nr[v] + ncj), 0.f);
            zv[v] = fmaxf(fmaf(2.f * dsq, ir[v] * ic, 1.f), 1.f + EPSF);
        }

        // diagonal fixup: uniform branch, taken in exactly one tile per wave
        if (ct == diag_ct) {
#pragma unroll
            for (int v = 0; v < 8; ++v)
                if (8 * hiH + v == l15) zv[v] = INFINITY;
        }

        bool anyIns = false;
#pragma unroll
        for (int v = 0; v < 8; ++v) anyIns |= (zv[v] < tz[v][TOPK - 1]);

        // wave-uniform skip of the whole insertion chain (s_cbranch on ballot)
        if (__builtin_amdgcn_ballot_w32(anyIns)) {
#pragma unroll
            for (int v = 0; v < 8; ++v) {
                if (zv[v] < tz[v][TOPK - 1]) {
                    tz[v][TOPK - 1] = zv[v];
                    ti[v][TOPK - 1] = (unsigned)col;
#pragma unroll
                    for (int j = TOPK - 1; j > 0; --j) {
                        if (tz[v][j] < tz[v][j - 1]) {
                            float    tf = tz[v][j]; tz[v][j] = tz[v][j - 1]; tz[v][j - 1] = tf;
                            unsigned tu = ti[v][j]; ti[v][j] = ti[v][j - 1]; ti[v][j - 1] = tu;
                        }
                    }
                }
            }
        }
    }

    // ---- dump per-lane candidates to LDS: [rowLocal][colLane][5]
#pragma unroll
    for (int v = 0; v < 8; ++v) {
        int rl = wave * 16 + 8 * hiH + v;
#pragma unroll
        for (int j = 0; j < TOPK; ++j) {
            s_z[rl][l15][j] = tz[v][j];
            s_i[rl][l15][j] = ti[v][j];
        }
    }
    __syncthreads();

    // ---- finalize: one thread per row (64 rows per block)
    if (tid < 64) {
        const int rl  = tid;
        const int row = blockIdx.x * 64 + rl;

        // exact top-5 from 16x5 candidates
        float    bz[TOPK];
        unsigned bi[TOPK];
#pragma unroll
        for (int j = 0; j < TOPK; ++j) { bz[j] = INFINITY; bi[j] = 0u; }
        for (int l = 0; l < 16; ++l) {
#pragma unroll
            for (int j = 0; j < TOPK; ++j) {
                float    z  = s_z[rl][l][j];
                unsigned id = s_i[rl][l][j];
                if (z < bz[TOPK - 1]) {
                    bz[TOPK - 1] = z; bi[TOPK - 1] = id;
#pragma unroll
                    for (int q = TOPK - 1; q > 0; --q) {
                        if (bz[q] < bz[q - 1]) {
                            float    tf = bz[q]; bz[q] = bz[q - 1]; bz[q - 1] = tf;
                            unsigned tu = bi[q]; bi[q] = bi[q - 1]; bi[q - 1] = tu;
                        }
                    }
                }
            }
        }

        // positive term: logsumexp(-acosh(z)/tau) over top-5 (acosh monotone,
        // so top-k on z == top-k on distance; transcendentals only here)
        float pm = -INFINITY;
        float pa[TOPK];
#pragma unroll
        for (int j = 0; j < TOPK; ++j) {
            pa[j] = -acoshf(bz[j]) / TAU_F;
            pm = fmaxf(pm, pa[j]);
        }
        float ps = 0.f;
#pragma unroll
        for (int j = 0; j < TOPK; ++j) ps += expf(pa[j] - pm);
        float pos_term = pm + logf(ps);

        // sorted (ascending) excluded indices for pool-skip mapping
        unsigned si5[TOPK];
#pragma unroll
        for (int j = 0; j < TOPK; ++j) si5[j] = bi[j];
#pragma unroll
        for (int a = 0; a < TOPK; ++a)
#pragma unroll
            for (int b = 0; b < TOPK - 1; ++b)
                if (si5[b] > si5[b + 1]) {
                    unsigned t = si5[b]; si5[b] = si5[b + 1]; si5[b + 1] = t;
                }

        const float nr_  = nsq[row];
        const float irow = inv[row];
        const v4f*  pr4  = (const v4f*)(x + (size_t)row * D_DIM);

        // negatives: threefry2x32(key=(0,42)), JAX split-counter layout,
        // streaming logsumexp; diagonal sample => exp(-inf) = 0
        float nm = -INFINITY, nsum = 0.f;
        const unsigned halfn = (unsigned)(B_N * NEGS) / 2u;
        for (int t = 0; t < NEGS; ++t) {
            unsigned fidx = (unsigned)row * NEGS + (unsigned)t;
            unsigned p0 = (fidx < halfn) ? fidx : (fidx - halfn);
            unsigned o0, o1;
            threefry2x32(0u, 42u, p0, p0 + halfn, o0, o1);
            unsigned bits = (fidx < halfn) ? o0 : o1;
            unsigned s    = bits % (unsigned)(B_N - TOPK);
            unsigned coln = s;
#pragma unroll
            for (int j = 0; j < TOPK; ++j) coln += (coln >= si5[j]) ? 1u : 0u;
            if ((int)coln == row) continue;

            const v4f* pc4 = (const v4f*)(x + (size_t)coln * D_DIM);
            float dot = 0.f;
#pragma unroll
            for (int c = 0; c < D_DIM / 4; ++c) {
                v4f a = pr4[c], b = pc4[c];
                dot = fmaf(a[0], b[0], dot);
                dot = fmaf(a[1], b[1], dot);
                dot = fmaf(a[2], b[2], dot);
                dot = fmaf(a[3], b[3], dot);
            }
            float ncn = nsq[coln];
            float dsq = fmaxf(nr_ + ncn - 2.f * dot, 0.f);
            float z   = fmaxf(fmaf(2.f * dsq, irow * inv[coln], 1.f), 1.f + EPSF);
            float a   = -acoshf(z) / TAU_F;
            if (a <= nm) {
                nsum += expf(a - nm);
            } else {
                nsum = nsum * expf(nm - a) + 1.f;
                nm = a;
            }
        }
        float neg_term = (nsum > 0.f) ? (nm + logf(nsum)) : -INFINITY;
        loss[row] = -(pos_term - neg_term);
    }
}

// ---------------------------------------------------------------------------
// Kernel 3: deterministic mean reduction of 8192 per-row losses
// ---------------------------------------------------------------------------
__global__ void reduce_kernel(const float* __restrict__ loss, float* __restrict__ out) {
    __shared__ float sm[256];
    float s = 0.f;
    for (int i = threadIdx.x; i < B_N; i += 256) s += loss[i];
    sm[threadIdx.x] = s;
    __syncthreads();
    for (int st = 128; st > 0; st >>= 1) {
        if ((int)threadIdx.x < st) sm[threadIdx.x] += sm[threadIdx.x + st];
        __syncthreads();
    }
    if (threadIdx.x == 0) out[0] = sm[0] / (float)B_N;
}

// ---------------------------------------------------------------------------
extern "C" void kernel_launch(void* const* d_in, const int* in_sizes, int n_in,
                              void* d_out, int out_size, void* d_ws, size_t ws_size,
                              hipStream_t stream) {
    const float* x = (const float*)d_in[0];

    // workspace layout (2.19 MB total):
    //   xh   : 8192*64 f16  (1 MB)   hi part of x
    //   xl   : 8192*64 f16  (1 MB)   lo (residual) part of x
    //   nsq  : 8192 f32
    //   inv  : 8192 f32              1 / (1 - clamp(||x||^2))
    //   loss : 8192 f32
    _Float16* xh  = (_Float16*)d_ws;
    _Float16* xl  = xh + (size_t)B_N * D_DIM;
    float*    nsq = (float*)(xl + (size_t)B_N * D_DIM);
    float*    inv = nsq + B_N;
    float*    loss = inv + B_N;

    prep_kernel<<<B_N / 256, 256, 0, stream>>>(x, xh, xl, nsq, inv);
    fused_kernel<<<B_N / 64, 128, 0, stream>>>(x, xh, xl, nsq, inv, loss);
    reduce_kernel<<<1, 256, 0, stream>>>(loss, (float*)d_out);
}
